// Tno2D_15625091023463
// MI455X (gfx1250) — matrix operand order repaired and checked by reading, not compiled, not run on hardware
//
#include <hip/hip_runtime.h>

typedef __attribute__((ext_vector_type(16))) __bf16 v16bf;
typedef __attribute__((ext_vector_type(8)))  __bf16 v8bf;
typedef __attribute__((ext_vector_type(8)))  float  v8f;
typedef __attribute__((ext_vector_type(4)))  int    v4i;

#define Hh 8
#define Nn 64
#define Mm 64
#define Dd 64
#define PD 48
#define Ll 3
#define Bb 8
#define N1 127
#define RSTRIDE 128   // padded row stride of rev plane (elements)

#if __has_builtin(__builtin_amdgcn_global_load_async_to_lds_b128) && \
    __has_builtin(__builtin_amdgcn_s_wait_asynccnt)
#define USE_ASYNC_LDS 1
// global (AS1) source and LDS (AS3) destination, exact pointee type int4
#define ASYNC_GPTR(p) ((__attribute__((address_space(1))) v4i*)(p))
#define ASYNC_SPTR(p) ((__attribute__((address_space(3))) v4i*)(p))
#else
#define USE_ASYNC_LDS 0
#endif

// ---------------------------------------------------------------------------
// Kernel 1: transpose x [B,H,N,M,D] f32 -> xt [B,H,D,N*M] bf16 (coalesced both ways)
// ---------------------------------------------------------------------------
__global__ __launch_bounds__(256) void xpose_kernel(const float* __restrict__ x,
                                                    __bf16* __restrict__ xt) {
  __shared__ __bf16 tile[64][65];
  const int il0   = blockIdx.x * 64;            // tile along flattened (i,l) = 4096
  const int plane = blockIdx.y;                 // b*8 + h
  const float* src = x  + (size_t)plane * (4096 * 64);
  __bf16*      dst = xt + (size_t)plane * (64 * 4096);
  const int t = threadIdx.x;
#pragma unroll
  for (int rep = 0; rep < 16; ++rep) {
    int idx = rep * 256 + t;
    int r = idx >> 6, c = idx & 63;             // consecutive t -> consecutive d: coalesced
    tile[r][c] = (__bf16)src[(size_t)(il0 + r) * 64 + c];
  }
  __syncthreads();
#pragma unroll
  for (int rep = 0; rep < 16; ++rep) {
    int idx = rep * 256 + t;
    int dd = idx >> 6, j = idx & 63;            // consecutive t -> consecutive il: coalesced
    dst[(size_t)dd * 4096 + il0 + j] = tile[j][dd];
  }
}

// ---------------------------------------------------------------------------
// Kernel 2: position MLP -> reversed, decayed coef table
//   rev[h][d][arow][s] = gamma^(|rn|+|rm|) * mlp(rn, rm)[h*64+d]
//   arow = rn+63, s = 63-rm  (reversal makes the Toeplitz B operand contiguous)
// ---------------------------------------------------------------------------
__global__ __launch_bounds__(128) void coef_kernel(
    const float* __restrict__ w_pos,   const float* __restrict__ b_pos,
    const float* __restrict__ w_layers,const float* __restrict__ b_layers,
    const float* __restrict__ w_out,   const float* __restrict__ b_out,
    __bf16* __restrict__ rev) {
  const int idx = blockIdx.x * 128 + threadIdx.x;
  if (idx >= N1 * N1) return;
  const int tn = idx / N1, tm = idx % N1;
  const int rn = (tn < 64) ? tn : tn - 127;     // signed row offset
  const int rm = (tm < 64) ? tm : tm - 127;     // signed col offset

  float h[PD], g[PD];
  const float fr = (float)rn, fc = (float)rm;
#pragma unroll
  for (int p = 0; p < PD; ++p)
    h[p] = fr * w_pos[p] + fc * w_pos[PD + p] + b_pos[p];

  for (int layer = 0; layer < Ll; ++layer) {
    float ss = 0.f;
#pragma unroll
    for (int p = 0; p < PD; ++p) ss += h[p] * h[p];
    float inv = 1.f / fmaxf(sqrtf(ss * (1.f / PD)), 1e-8f);
#pragma unroll
    for (int p = 0; p < PD; ++p) g[p] = fmaxf(h[p] * inv, 0.f);
    const float* w = w_layers + (size_t)layer * PD * PD;
    for (int q = 0; q < PD; ++q) {
      float acc = b_layers[layer * PD + q];
#pragma unroll
      for (int p = 0; p < PD; ++p) acc += g[p] * w[p * PD + q];
      h[q] = acc;
    }
  }
  {
    float ss = 0.f;
#pragma unroll
    for (int p = 0; p < PD; ++p) ss += h[p] * h[p];
    float inv = 1.f / fmaxf(sqrtf(ss * (1.f / PD)), 1e-8f);
#pragma unroll
    for (int p = 0; p < PD; ++p) g[p] = fmaxf(h[p] * inv, 0.f);
  }
  const int an = (rn < 0) ? -rn : rn;
  const int am = (rm < 0) ? -rm : rm;
  const float decay = __powf(0.999f, (float)(an + am));
  const int arow = rn + 63;
  const int s    = 63 - rm;
  for (int v = 0; v < Hh * Dd; ++v) {           // all lanes share v -> uniform weight reads
    float acc = b_out[v];
#pragma unroll
    for (int p = 0; p < PD; ++p) acc += g[p] * w_out[p * (Hh * Dd) + v];
    rev[((size_t)v * N1 + arow) * RSTRIDE + s] = (__bf16)(acc * decay);
  }
}

// ---------------------------------------------------------------------------
// Kernel 3: Toeplitz contraction via v_wmma_f32_16x16x32_bf16
//   One workgroup per (h,d,b); 16 waves = 4 i-blocks x 4 j-blocks.
//   r-loop padded to 128; processed in groups of 2 per barrier phase.
// ---------------------------------------------------------------------------
__global__ __launch_bounds__(512) void tno_conv_kernel(
    const __bf16* __restrict__ xt, const __bf16* __restrict__ rev,
    float* __restrict__ out) {
  __shared__ __bf16 xlds[64 * 64];                          // 8 KB   x plane (bf16)
  __shared__ __bf16 revlds[128 * RSTRIDE];                  // 32 KB  coef plane (row 127 pad)
  __shared__ __align__(32) __bf16 frag[2][2][8][512];       // 32 KB  prebuilt B fragments

  const int d = blockIdx.x, h = blockIdx.y, b = blockIdx.z;
  const int t = threadIdx.x;

  // ---- stage x plane and rev plane into LDS ----
  const v8bf* xsrc = (const v8bf*)(xt + ((size_t)((b * Hh + h) * Dd + d)) * 4096);
  const v8bf* rsrc = (const v8bf*)(rev + ((size_t)(h * Dd + d)) * (N1 * RSTRIDE));
#if USE_ASYNC_LDS
  __builtin_amdgcn_global_load_async_to_lds_b128(
      ASYNC_GPTR(xsrc + t), ASYNC_SPTR((v8bf*)xlds + t), 0, 0);
  for (int c = t; c < (N1 * RSTRIDE) / 8; c += 512)
    __builtin_amdgcn_global_load_async_to_lds_b128(
        ASYNC_GPTR(rsrc + c), ASYNC_SPTR((v8bf*)revlds + c), 0, 0);
  __builtin_amdgcn_s_wait_asynccnt(0);
#else
  ((v8bf*)xlds)[t] = xsrc[t];
  for (int c = t; c < (N1 * RSTRIDE) / 8; c += 512) ((v8bf*)revlds)[c] = rsrc[c];
#endif
  __syncthreads();

  const int lane = t & 31;
  const int w    = t >> 5;
  const int i0 = __builtin_amdgcn_readfirstlane(16 * (w & 3));
  const int j0 = __builtin_amdgcn_readfirstlane(16 * (w >> 2));

  // ---- builder assignment: frag f = (l0/32)*4 + j0/16, WMMA-native lane-major ----
  const int bfid = t >> 6;                // fragment id 0..7
  const int bu   = t & 63;
  const int bln  = bu >> 1;               // target lane 0..31
  const int be0  = (bu & 1) * 8;          // element chunk 0 or 8
  const int bj0  = (bfid & 3) * 16;
  const int bl0  = (bfid >> 2) * 32;
  const int bn   = bln & 15;              // N index of that lane
  const int bk   = ((bln >> 4) * 8) + (bu & 1) * 16;     // K base for that lane/chunk
  const int sbase = 63 + bl0 - bj0 - bn + bk;   // s = sbase + e, always in [0,126]

  // build both fragments of group 0 (ar = 0,1)
#pragma unroll
  for (int rr = 0; rr < 2; ++rr) {
    const __bf16* rrow = &revlds[(0 + rr) * RSTRIDE + sbase];
    v8bf tmp;
#pragma unroll
    for (int e = 0; e < 8; ++e) tmp[e] = rrow[e];
    *(v8bf*)&frag[0][rr][bfid][bln * 16 + be0] = tmp;
  }
  __syncthreads();

  v8f acc = {};
  const int mrow  = lane & 15;            // A-matrix row for this lane
  const int half8 = (lane >> 4) * 8;      // K sub-chunk select per A layout

  for (int g = 0; g < 64; ++g) {
    const int cur = g & 1;
#pragma unroll
    for (int rr = 0; rr < 2; ++rr) {
      const int ar = 2 * g + rr;
      const int rn = ar - 63;                       // ar==127 (rn=64) skipped by all waves
      if (i0 + 15 - rn >= 0 && i0 - rn <= 63) {     // wave-uniform skip
        const int  ip    = i0 + mrow - rn;
        const bool valid = (ip >= 0) && (ip < 64);
#pragma unroll
        for (int c2 = 0; c2 < 2; ++c2) {            // K=64 as two K=32 WMMAs
          const int l0 = c2 * 32;
          v16bf a = {};
          if (valid) {
            const __bf16* xr = &xlds[ip * 64 + l0 + half8];   // 16B aligned
            v8bf lo = *(const v8bf*)xr;                       // K = 8*half ..
            v8bf hi = *(const v8bf*)(xr + 16);                // K = 16+8*half ..
            a = __builtin_shufflevector(lo, hi, 0, 1, 2, 3, 4, 5, 6, 7,
                                        8, 9, 10, 11, 12, 13, 14, 15);
          }
          const int fid = c2 * 4 + (j0 >> 4);
          v16bf bm = *(const v16bf*)&frag[cur][rr][fid][lane * 16]; // 2x ds_load_b128
          acc = __builtin_amdgcn_wmma_f32_16x16x32_bf16(
              false, a, false, bm, (short)0, acc, false, false);
        }
      }
    }
    if (g + 1 < 64) {                               // double-buffered build of next group
      const int nxt = (g + 1) & 1;
#pragma unroll
      for (int rr = 0; rr < 2; ++rr) {
        const __bf16* rrow = &revlds[(2 * (g + 1) + rr) * RSTRIDE + sbase];
        v8bf tmp;
#pragma unroll
        for (int e = 0; e < 8; ++e) tmp[e] = rrow[e];
        *(v8bf*)&frag[nxt][rr][bfid][bln * 16 + be0] = tmp;
      }
    }
    __syncthreads();
  }

  // ---- store D tile per documented 16x16 f32 C/D layout ----
  const int n     = lane & 15;
  const int mbase = 8 * (lane >> 4);
  size_t obase = ((((size_t)(b * Hh + h)) * 64 + i0 + mbase) * 64 + (j0 + n)) * 64 + d;
#pragma unroll
  for (int p = 0; p < 8; ++p)
    out[obase + (size_t)p * (64 * 64)] = acc[p];
}

// ---------------------------------------------------------------------------
extern "C" void kernel_launch(void* const* d_in, const int* in_sizes, int n_in,
                              void* d_out, int out_size, void* d_ws, size_t ws_size,
                              hipStream_t stream) {
  (void)in_sizes; (void)n_in; (void)out_size; (void)ws_size;
  const float* x        = (const float*)d_in[0];
  const float* w_pos    = (const float*)d_in[1];
  const float* b_pos    = (const float*)d_in[2];
  const float* w_layers = (const float*)d_in[3];
  const float* b_layers = (const float*)d_in[4];
  const float* w_out    = (const float*)d_in[5];
  const float* b_out    = (const float*)d_in[6];
  float* out = (float*)d_out;

  __bf16* xt  = (__bf16*)d_ws;                                  // 33.5 MB
  __bf16* rev = (__bf16*)((char*)d_ws +
                 (size_t)Bb * Hh * Dd * Nn * Mm * sizeof(__bf16)); // 16.6 MB

  xpose_kernel<<<dim3(64, 64), 256, 0, stream>>>(x, xt);
  coef_kernel<<<(N1 * N1 + 127) / 128, 128, 0, stream>>>(
      w_pos, b_pos, w_layers, b_layers, w_out, b_out, rev);
  tno_conv_kernel<<<dim3(Dd, Hh, Bb), 512, 0, stream>>>(xt, rev, out);
}
